// GNN_31877247271117
// MI455X (gfx1250) — matrix operand order repaired and checked by reading, not compiled, not run on hardware
//
#include <hip/hip_runtime.h>

#define KNBR 16
#define DH   128

typedef __attribute__((ext_vector_type(8)))  float          v8f;
typedef __attribute__((ext_vector_type(16))) __bf16         v16bf;
typedef __attribute__((ext_vector_type(2)))  __bf16         v2bf;
typedef __attribute__((ext_vector_type(8)))  unsigned short v8u;
typedef __attribute__((ext_vector_type(16))) unsigned short v16u;

union ABits { v8u h[2]; v16bf v; };
union BBits { v16u u;   v16bf v; };

__device__ __forceinline__ unsigned short f2bf(float x) {
  unsigned int u = __float_as_uint(x);
  u += 0x7FFFu + ((u >> 16) & 1u);          // round-to-nearest-even
  return (unsigned short)(u >> 16);
}

// packed f32x2 -> bf16x2 (one v_cvt_pk_bf16_f32 when available)
__device__ __forceinline__ unsigned int f2bf2(float lo, float hi) {
#if __has_builtin(__builtin_amdgcn_cvt_pk_bf16_f32)
  v2bf r = __builtin_amdgcn_cvt_pk_bf16_f32(lo, hi);
  return __builtin_bit_cast(unsigned int, r);
#else
  return (unsigned int)f2bf(lo) | ((unsigned int)f2bf(hi) << 16);
#endif
}

// native v_tanh_f32 on gfx1250 (TRANS op) when available
__device__ __forceinline__ float fast_tanh(float x) {
#if __has_builtin(__builtin_amdgcn_tanhf)
  return __builtin_amdgcn_tanhf(x);
#else
  return tanhf(x);
#endif
}
__device__ __forceinline__ float sigm(float x) {
  return 0.5f * fast_tanh(0.5f * x) + 0.5f;   // exact identity
}

__device__ __forceinline__ v8f wmma_bf16(v16bf a, v16bf b, v8f c) {
  // D = A(16x32 bf16) * B(32x16 bf16) + C(16x16 f32)
  return __builtin_amdgcn_wmma_f32_16x16x32_bf16(false, a, false, b, (short)0, c,
                                                 false, false);
}

// A operand (16x32 bf16 tile) from LDS row-major [16][136] (contiguous k order, same as loadB):
// lane m=l&15; halves 0..15 = K base+0..15 contiguous,
// base = (l>=16 ? 16 : 0) within the 32-wide K chunk kc.
__device__ __forceinline__ v16bf loadA(const unsigned short (*S)[136], int lane, int kc) {
  int m    = lane & 15;
  int koff = kc * 32 + ((lane >> 4) << 4);
  ABits a;
  a.h[0] = *(const v8u*)&S[m][koff];
  a.h[1] = *(const v8u*)&S[m][koff + 8];
  return a.v;
}

// B operand (32x16 bf16 tile) from a row-major bf16 weight matrix W[row][0..127]:
// lane holds column n=l&15 with K = (l>=16?16:0)+0..15 contiguous -> one 32B load.
// Loop-invariant across timesteps -> compiler keeps these resident in the
// extended (1024) VGPR file (weight-stationary persistent-LSTM).
__device__ __forceinline__ v16bf loadB(const unsigned short* __restrict__ W,
                                       int row, int kc, int lane) {
  BBits b;
  b.u = *(const v16u*)(W + (size_t)row * DH + kc * 32 + ((lane >> 4) << 4));
  return b.v;
}

// ---------------- prep kernels (weights f32 -> bf16, optional transpose) ---------

__global__ void cvt_bf16_kernel(const float* __restrict__ s,
                                unsigned short* __restrict__ d, int n) {
  int i = blockIdx.x * blockDim.x + threadIdx.x;
  if (i < n) d[i] = f2bf(s[i]);
}

__global__ void transpose_bf16_kernel(const float* __restrict__ s,
                                      unsigned short* __restrict__ d, int R, int C) {
  int i = blockIdx.x * blockDim.x + threadIdx.x;
  if (i < R * C) { int r = i / C, c = i % C; d[(size_t)c * R + r] = f2bf(s[i]); }
}

// ---------------- fused SAGE-LSTM layer -----------------------------------------
// One workgroup (4 wave32) = 16 nodes.  Wave g computes gate g (i,f,g,o).
// Double-buffered gather: step t+1's neighbor rows are fetched into registers
// while step t's WMMAs run, committed to the alternate LDS buffer after the
// elementwise phase.  2 barriers per timestep.

template<bool IN_BF16, bool OUT_BF16, bool RELU, int OUTC>
__global__ __launch_bounds__(128)
void sage_lstm_layer(const void* __restrict__ in_feat,
                     const int*  __restrict__ nbrs,
                     const unsigned short* __restrict__ Wih,      // [512][128] bf16
                     const unsigned short* __restrict__ Whh,      // [512][128] bf16
                     const float* __restrict__ bih,               // [512]
                     const float* __restrict__ bhh,               // [512]
                     const unsigned short* __restrict__ WselfT,   // [OUTC][128] bf16
                     const unsigned short* __restrict__ WneighT,  // [OUTC][128] bf16
                     const float* __restrict__ bias,              // [OUTC]
                     void* __restrict__ out) {
  __shared__ unsigned short sAx[2][16][136];   // x_t tiles, bf16 (double buffered)
  __shared__ unsigned short sAh[16][136];      // h tile,  bf16
  __shared__ float          sC [16][DH];       // cell state
  __shared__ float          sG [4][16][DH+4];  // gate blocks i/f/g/o

  const int tid  = threadIdx.x;
  const int wave = tid >> 5;
  const int lane = tid & 31;
  const int node_base = blockIdx.x * 16;       // N = 100000 = 6250*16, no guard

  // h0 = c0 = 0
  for (int m = 0; m < 16; ++m) { sAh[m][tid] = 0; sC[m][tid] = 0.0f; }

  // per-column fused biases (column owned by this thread in elementwise phase)
  const float bsi = bih[tid]        + bhh[tid];
  const float bsf = bih[DH + tid]   + bhh[DH + tid];
  const float bsg = bih[2*DH + tid] + bhh[2*DH + tid];
  const float bso = bih[3*DH + tid] + bhh[3*DH + tid];

  const int lm   = tid >> 3;   // gather-loader: row 0..15
  const int lseg = tid & 7;    // 16-col segment
  const int* __restrict__ myN = nbrs + (size_t)(node_base + lm) * KNBR;
  const unsigned short* __restrict__ inB = (const unsigned short*)in_feat;
  const float*          __restrict__ inF = (const float*)in_feat;

  float4 pf[4];   // prefetch regs (f32 input)
  v8u    pb[2];   // prefetch regs (bf16 input)

  auto prefetch = [&](int t) {
    int idx = myN[t];
    if (IN_BF16) {
      const v8u* s = (const v8u*)(inB + (size_t)idx * DH + lseg * 16);
      pb[0] = s[0]; pb[1] = s[1];
    } else {
      const float4* s = (const float4*)(inF + (size_t)idx * DH + lseg * 16);
      pf[0] = s[0]; pf[1] = s[1]; pf[2] = s[2]; pf[3] = s[3];
    }
  };
  auto commit = [&](int buf) {
    if (IN_BF16) {
      *(v8u*)&sAx[buf][lm][lseg * 16]     = pb[0];
      *(v8u*)&sAx[buf][lm][lseg * 16 + 8] = pb[1];
    } else {
      #pragma unroll
      for (int q = 0; q < 4; ++q) {
        float4 v = pf[q];
        *(unsigned int*)&sAx[buf][lm][lseg * 16 + q * 4]     = f2bf2(v.x, v.y);
        *(unsigned int*)&sAx[buf][lm][lseg * 16 + q * 4 + 2] = f2bf2(v.z, v.w);
      }
    }
  };

  prefetch(0);
  commit(0);
  __syncthreads();                       // sAx[0], sAh, sC visible

  for (int t = 0; t < KNBR; ++t) {
    const int buf = t & 1;

    // issue next step's gather now; consumed after the elementwise phase
    if (t + 1 < KNBR) prefetch(t + 1);

    // ---- this wave's gate block: 16x128, K = 128(x) + 128(h) ----
    v16bf ax[4], ah[4];
    #pragma unroll
    for (int kc = 0; kc < 4; ++kc) {
      ax[kc] = loadA(sAx[buf], lane, kc);
      ah[kc] = loadA(sAh,      lane, kc);
    }

    v8f acc[8];
    #pragma unroll
    for (int ct = 0; ct < 8; ++ct) { v8f z = {}; acc[ct] = z; }

    #pragma unroll
    for (int ct = 0; ct < 8; ++ct) {
      int wrow = wave * DH + ct * 16 + (lane & 15);   // row of W_ih / W_hh
      #pragma unroll
      for (int kc = 0; kc < 4; ++kc) acc[ct] = wmma_bf16(ax[kc], loadB(Wih, wrow, kc, lane), acc[ct]);
      #pragma unroll
      for (int kc = 0; kc < 4; ++kc) acc[ct] = wmma_bf16(ah[kc], loadB(Whh, wrow, kc, lane), acc[ct]);
    }

    // spill gate block to LDS (C layout: vgpr r -> row r+8*(lane>=16), col lane&15)
    #pragma unroll
    for (int ct = 0; ct < 8; ++ct) {
      int col = ct * 16 + (lane & 15);
      int rb  = (lane >> 4) << 3;
      #pragma unroll
      for (int r = 0; r < 8; ++r) sG[wave][rb + r][col] = acc[ct][r];
    }
    __syncthreads();                     // gates visible to elementwise

    // ---- elementwise LSTM cell: thread owns column `tid`, all 16 rows ----
    #pragma unroll
    for (int m = 0; m < 16; ++m) {
      float gi = sigm     (sG[0][m][tid] + bsi);
      float gf = sigm     (sG[1][m][tid] + bsf);
      float gg = fast_tanh(sG[2][m][tid] + bsg);
      float go = sigm     (sG[3][m][tid] + bso);
      float c  = gf * sC[m][tid] + gi * gg;
      sC[m][tid]  = c;
      sAh[m][tid] = f2bf(go * fast_tanh(c));   // h_t, next step's A operand
    }

    // commit prefetched x_{t+1} into the alternate buffer
    if (t + 1 < KNBR) commit(buf ^ 1);
    __syncthreads();                     // orders sAh + sAx[buf^1] writes
  }

  // ---- output linear: out = self@W_selfT' + h_final@W_neighT' + bias ----
  {
    int node = node_base + lm;
    if (IN_BF16) {
      const v8u* src = (const v8u*)(inB + (size_t)node * DH + lseg * 16);
      *(v8u*)&sAx[0][lm][lseg * 16]     = src[0];
      *(v8u*)&sAx[0][lm][lseg * 16 + 8] = src[1];
    } else {
      const float4* src = (const float4*)(inF + (size_t)node * DH + lseg * 16);
      #pragma unroll
      for (int q = 0; q < 4; ++q) {
        float4 v = src[q];
        *(unsigned int*)&sAx[0][lm][lseg * 16 + q * 4]     = f2bf2(v.x, v.y);
        *(unsigned int*)&sAx[0][lm][lseg * 16 + q * 4 + 2] = f2bf2(v.z, v.w);
      }
    }
  }
  __syncthreads();

  constexpr int TILES = OUTC / 16;
  for (int tile = wave; tile < TILES; tile += 4) {
    v8f acc = {};
    int brow = tile * 16 + (lane & 15);
    #pragma unroll
    for (int kc = 0; kc < 4; ++kc) acc = wmma_bf16(loadA(sAx[0], lane, kc), loadB(WselfT,  brow, kc, lane), acc);
    #pragma unroll
    for (int kc = 0; kc < 4; ++kc) acc = wmma_bf16(loadA(sAh,    lane, kc), loadB(WneighT, brow, kc, lane), acc);

    int   col = tile * 16 + (lane & 15);
    float bv  = bias[col];
    int   rb  = (lane >> 4) << 3;
    #pragma unroll
    for (int r = 0; r < 8; ++r) {
      float v = acc[r] + bv;
      if (RELU) v = fmaxf(v, 0.0f);
      size_t o = (size_t)(node_base + rb + r) * OUTC + col;
      if (OUT_BF16) ((unsigned short*)out)[o] = f2bf(v);
      else          ((float*)out)[o] = v;
    }
  }
}

// ---------------- final neighbor sum: out[n,:] = sum_t h2[nbrs[n,t],:] ----------

__global__ void nbr_sum_kernel(const float* __restrict__ h2,
                               const int* __restrict__ nbrs,
                               float* __restrict__ out, int n) {
  int i = blockIdx.x * blockDim.x + threadIdx.x;
  if (i >= n * 16) return;
  int nd = i >> 4, c = i & 15;
  const int* nb = nbrs + (size_t)nd * KNBR;
  float s = 0.0f;
  #pragma unroll
  for (int t = 0; t < KNBR; ++t) s += h2[(size_t)nb[t] * 16 + c];
  out[i] = s;
}

// ---------------- host side -----------------------------------------------------

extern "C" void kernel_launch(void* const* d_in, const int* in_sizes, int n_in,
                              void* d_out, int out_size, void* d_ws, size_t ws_size,
                              hipStream_t stream) {
  const int N = in_sizes[0] / DH;   // 100000

  const float* feat     = (const float*)d_in[0];
  const int*   nbrs     = (const int*)  d_in[1];
  const float* W_ih1    = (const float*)d_in[2];
  const float* W_hh1    = (const float*)d_in[3];
  const float* b_ih1    = (const float*)d_in[4];
  const float* b_hh1    = (const float*)d_in[5];
  const float* W_self1  = (const float*)d_in[6];
  const float* W_neigh1 = (const float*)d_in[7];
  const float* b1       = (const float*)d_in[8];
  const float* W_ih2    = (const float*)d_in[9];
  const float* W_hh2    = (const float*)d_in[10];
  const float* b_ih2    = (const float*)d_in[11];
  const float* b_hh2    = (const float*)d_in[12];
  const float* W_self2  = (const float*)d_in[13];
  const float* W_neigh2 = (const float*)d_in[14];
  const float* b2       = (const float*)d_in[15];

  // workspace carve-up (256B aligned)
  char*  w   = (char*)d_ws;
  size_t off = 0;
  auto take = [&](size_t bytes) -> void* {
    void* p = w + off;
    off += (bytes + 255) & ~(size_t)255;
    return p;
  };
  unsigned short* Wih1b = (unsigned short*)take((size_t)4 * DH * DH * 2);
  unsigned short* Whh1b = (unsigned short*)take((size_t)4 * DH * DH * 2);
  unsigned short* Wst1  = (unsigned short*)take((size_t)DH * DH * 2);
  unsigned short* Wnt1  = (unsigned short*)take((size_t)DH * DH * 2);
  unsigned short* Wih2b = (unsigned short*)take((size_t)4 * DH * DH * 2);
  unsigned short* Whh2b = (unsigned short*)take((size_t)4 * DH * DH * 2);
  unsigned short* Wst2  = (unsigned short*)take((size_t)16 * DH * 2);
  unsigned short* Wnt2  = (unsigned short*)take((size_t)16 * DH * 2);
  unsigned short* h1    = (unsigned short*)take((size_t)N * DH * 2);   // layer-1 out, bf16
  float*          h2    = (float*)         take((size_t)N * 16 * 4);   // layer-2 out, f32

  // weight prep (tiny)
  const int nw = 4 * DH * DH;   // 65536
  cvt_bf16_kernel<<<(nw + 255) / 256, 256, 0, stream>>>(W_ih1, Wih1b, nw);
  cvt_bf16_kernel<<<(nw + 255) / 256, 256, 0, stream>>>(W_hh1, Whh1b, nw);
  cvt_bf16_kernel<<<(nw + 255) / 256, 256, 0, stream>>>(W_ih2, Wih2b, nw);
  cvt_bf16_kernel<<<(nw + 255) / 256, 256, 0, stream>>>(W_hh2, Whh2b, nw);
  transpose_bf16_kernel<<<(DH * DH + 255) / 256, 256, 0, stream>>>(W_self1,  Wst1, DH, DH);
  transpose_bf16_kernel<<<(DH * DH + 255) / 256, 256, 0, stream>>>(W_neigh1, Wnt1, DH, DH);
  transpose_bf16_kernel<<<(DH * 16 + 255) / 256, 256, 0, stream>>>(W_self2,  Wst2, DH, 16);
  transpose_bf16_kernel<<<(DH * 16 + 255) / 256, 256, 0, stream>>>(W_neigh2, Wnt2, DH, 16);

  const int blocks = N / 16;    // 6250

  // layer 1: f32 feat in, bf16 [N,128] out, ReLU
  sage_lstm_layer<false, true, true, DH><<<blocks, 128, 0, stream>>>(
      (const void*)feat, nbrs, Wih1b, Whh1b, b_ih1, b_hh1, Wst1, Wnt1, b1, (void*)h1);

  // layer 2: bf16 h1 in, f32 [N,16] out, no ReLU
  sage_lstm_layer<true, false, false, 16><<<blocks, 128, 0, stream>>>(
      (const void*)h1, nbrs, Wih2b, Whh2b, b_ih2, b_hh2, Wst2, Wnt2, b2, (void*)h2);

  // final neighbor sum -> d_out [N,16] f32
  nbr_sum_kernel<<<(N * 16 + 255) / 256, 256, 0, stream>>>(h2, nbrs, (float*)d_out, N);

  (void)n_in; (void)out_size; (void)ws_size;
}
